// AttentionLayer_16080357556498
// MI455X (gfx1250) — compile-verified, matrix operand order checked by reading, not run on hardware
//
#include <hip/hip_runtime.h>
#include <math.h>

#define S_ 2048
#define H_ 1024
#define A_ 64
#define B_ 16

typedef __attribute__((ext_vector_type(16))) __bf16 v16bf;
typedef __attribute__((ext_vector_type(8)))  __bf16 v8bf;
typedef __attribute__((ext_vector_type(8)))  float  v8f;
typedef __attribute__((ext_vector_type(4)))  unsigned int v4u;
typedef __attribute__((ext_vector_type(8)))  int v8i;
typedef __attribute__((ext_vector_type(4)))  int v4i;

__device__ __forceinline__ __bf16 f2bf(float f) { return (__bf16)f; }
__device__ __forceinline__ unsigned short bf_bits(float f) {
  __bf16 h = (__bf16)f;
  return __builtin_bit_cast(unsigned short, h);
}

// ---------------------------------------------------------------------------
// Pack Wq/Wk (H x A, f32) into B-fragment-ordered bf16:
// idx = ((nt*32 + ks)*32 + lane)*16 + e
// lane<16: n=nt*16+lane, k runs {ks*32+0..7, ks*32+16..23}
// lane>=16: same n, k runs {ks*32+8..15, ks*32+24..31}
// ---------------------------------------------------------------------------
__global__ void __launch_bounds__(256)
prep_wfrag(const float* __restrict__ Wq, const float* __restrict__ Wk,
           unsigned short* __restrict__ WqB, unsigned short* __restrict__ WkB) {
  int idx = blockIdx.x * 256 + threadIdx.x;      // 0 .. 65535
  int e    = idx & 15;
  int lane = (idx >> 4) & 31;
  int ks   = (idx >> 9) & 31;
  int nt   = (idx >> 14) & 3;
  int n = nt * 16 + (lane & 15);
  int k = ks * 32 + ((lane >> 4) << 3) + ((e & 8) ? (16 + (e & 7)) : (e & 7));
  WqB[idx] = bf_bits(Wq[(size_t)k * A_ + n]);
  WkB[idx] = bf_bits(Wk[(size_t)k * A_ + n]);
}

// ---------------------------------------------------------------------------
// Q = x@Wq + bq ; K = x@Wk + bk   (bf16 WMMA, one wave per 16-row tile)
// ---------------------------------------------------------------------------
__global__ void __launch_bounds__(128)
proj_qk(const float* __restrict__ x, const float* __restrict__ bq,
        const float* __restrict__ bk,
        const unsigned short* __restrict__ WqB, const unsigned short* __restrict__ WkB,
        unsigned short* __restrict__ Qb, unsigned short* __restrict__ Kb) {
  const int lane = threadIdx.x & 31;
  const int wave = threadIdx.x >> 5;
  const int rowtile = blockIdx.x * 4 + wave;     // 0 .. 2047
  const int b  = rowtile >> 7;                   // / (S/16)
  const int s0 = (rowtile & 127) << 4;
  const int half = lane >> 4;
  const int ln   = lane & 15;

  v8f accQ[4] = {};
  v8f accK[4] = {};

  const float* xrow = x + ((size_t)(b * S_ + s0 + ln)) * H_;
  for (int ks = 0; ks < 32; ++ks) {
    const int kb = ks * 32 + half * 8;
    v16bf a;
#pragma unroll
    for (int i = 0; i < 8; ++i) {
      a[i]     = f2bf(xrow[kb + i]);
      a[8 + i] = f2bf(xrow[kb + 16 + i]);
    }
#pragma unroll
    for (int nt = 0; nt < 4; ++nt) {
      v16bf wq = *(const v16bf*)(WqB + ((size_t)((nt * 32 + ks) * 32 + lane) << 4));
      v16bf wk = *(const v16bf*)(WkB + ((size_t)((nt * 32 + ks) * 32 + lane) << 4));
      accQ[nt] = __builtin_amdgcn_wmma_f32_16x16x32_bf16(false, a, false, wq,
                                                         (short)0, accQ[nt], false, false);
      accK[nt] = __builtin_amdgcn_wmma_f32_16x16x32_bf16(false, a, false, wk,
                                                         (short)0, accK[nt], false, false);
    }
  }
  // D layout: lane holds (m = j + 8*half, n = ln) in VGPR j
#pragma unroll
  for (int nt = 0; nt < 4; ++nt) {
    const int col = nt * 16 + ln;
    const float biasq = bq[col], biask = bk[col];
#pragma unroll
    for (int j = 0; j < 8; ++j) {
      const int m = j + half * 8;
      size_t o = ((size_t)(b * S_ + s0 + m)) * A_ + col;
      Qb[o] = bf_bits(accQ[nt][j] + biasq);
      Kb[o] = bf_bits(accK[nt][j] + biask);
    }
  }
}

__device__ __forceinline__ void load_kfrag(const unsigned short* __restrict__ row,
                                           int kb, v16bf& f0, v16bf& f1) {
  v8bf lo0 = *(const v8bf*)(row + kb);
  v8bf hi0 = *(const v8bf*)(row + kb + 16);
  v8bf lo1 = *(const v8bf*)(row + kb + 32);
  v8bf hi1 = *(const v8bf*)(row + kb + 48);
#pragma unroll
  for (int i = 0; i < 8; ++i) {
    f0[i] = lo0[i]; f0[8 + i] = hi0[i];
    f1[i] = lo1[i]; f1[8 + i] = hi1[i];
  }
}

// ---------------------------------------------------------------------------
// TDM: stage a chunk of the K panel (CHUNK_ROWS x 64 bf16, row stride 64)
// into LDS. D# per CDNA5 ISA 8.3/8.4: group0 = {count=1, lds_addr,
// global_addr, type=2}; group1 = {data_size=2B, tensor_dim0=64,
// tensor_dim1=2048, tile_dim0=64, tile_dim1=CHUNK_ROWS, dim0_stride=64}.
// Groups 2/3 (and the extra group of the 6-arg builtin) zero: 2-D tensor.
// Tracked by TENSORcnt.
// ---------------------------------------------------------------------------
#define CHUNK_T    8                    // column tiles per chunk
#define CHUNK_ROWS (CHUNK_T * 16)       // 128 K-rows = 16 KB per buffer
#define NCHUNK     (S_ / CHUNK_ROWS)    // 16

__device__ __forceinline__ void tdm_load_chunk(const unsigned short* gsrc,
                                               unsigned lds_off) {
  unsigned long long ga = (unsigned long long)gsrc;
  v4u g0 = { 1u,                                   // count=1, user mode
             lds_off,                              // lds_addr
             (unsigned)(ga & 0xFFFFFFFFu),         // global_addr[31:0]
             (unsigned)(((ga >> 32) & 0x01FFFFFFu) | 0x80000000u) }; // [56:32] | type=2
  v8i g1 = { 0x00010000,                           // data_size = 1 (2 bytes)
             0x00400000,                           // tensor_dim0 = 64 (bits 79:48 lo)
             0x08000000,                           // tensor_dim1 = 2048 (bits 111:80 lo)
             0x00400000,                           // tile_dim0 = 64 (bits 127:112)
             CHUNK_ROWS,                           // tile_dim1
             64,                                   // tensor_dim0_stride = 64
             0, 0 };
  v4i gz4 = { 0, 0, 0, 0 };
  v8i gz8 = { 0, 0, 0, 0, 0, 0, 0, 0 };
  __builtin_amdgcn_tensor_load_to_lds(g0, g1, gz4, gz4, gz8, 0);
}

// ---------------------------------------------------------------------------
// scores = Q K^T / 8, column-masked softmax, accumulate column sums
//   w_all[b,t]  = sum_s attn[b,s,t]
//   w_msk[b,t]  = sum_s mask[b,s] * attn[b,s,t]
// One wave per 16 query rows; K panel staged in LDS by the TDM (double
// buffered, shared by all 8 waves of the block); two-pass online softmax.
// ---------------------------------------------------------------------------
__global__ void __launch_bounds__(256)
attn_colsum(const unsigned short* __restrict__ Qb, const unsigned short* __restrict__ Kb,
            const int* __restrict__ mask,
            float* __restrict__ w_all, float* __restrict__ w_msk) {
  __shared__ unsigned short ksh[2][CHUNK_ROWS * A_];   // 2 x 16 KB

  const int lane = threadIdx.x & 31;
  const int wave = threadIdx.x >> 5;
  const int rowtile = blockIdx.x * 8 + wave;     // 0 .. 2047 (same batch per block)
  const int b  = rowtile >> 7;
  const int s0 = (rowtile & 127) << 4;
  const int half = lane >> 4;
  const int ln   = lane & 15;
  const int kb   = half * 8;

  const unsigned short* kpanel = Kb + (size_t)b * S_ * A_;
  const unsigned lds0 = (unsigned)(unsigned long long)(const void*)&ksh[0][0];
  const unsigned lds1 = (unsigned)(unsigned long long)(const void*)&ksh[1][0];

  // A fragments of Q (16 x 64 bf16 -> two 16x32 fragments)
  const unsigned short* qrow = Qb + ((size_t)(b * S_ + s0 + ln)) * A_;
  v16bf aq0, aq1;
  load_kfrag(qrow, kb, aq0, aq1);

  const int* mrow = mask + (size_t)b * S_;
  float mr[8];
#pragma unroll
  for (int j = 0; j < 8; ++j) mr[j] = (float)mrow[s0 + j + half * 8];

  float mL[8], lL[8];
#pragma unroll
  for (int j = 0; j < 8; ++j) { mL[j] = -1e30f; lL[j] = 0.f; }

  // ---- pass 1: row max + sum of exp (online, per-lane partials) ----
  if (wave == 0) tdm_load_chunk(kpanel, lds0);
  for (int c = 0; c < NCHUNK; ++c) {
    __builtin_amdgcn_s_wait_tensorcnt(0);
    __syncthreads();                              // chunk c visible in buf c&1
    if (wave == 0 && c + 1 < NCHUNK)
      tdm_load_chunk(kpanel + (size_t)(c + 1) * CHUNK_ROWS * A_,
                     ((c + 1) & 1) ? lds1 : lds0);
    const unsigned short* kc = &ksh[c & 1][0];
    for (int tl = 0; tl < CHUNK_T; ++tl) {
      const int ct = c * CHUNK_T + tl;
      const bool cm = mrow[ct * 16 + ln] != 0;
      const unsigned short* krow = kc + (tl * 16 + ln) * A_;
      v16bf bk0, bk1;
      load_kfrag(krow, kb, bk0, bk1);
      v8f acc = {};
      acc = __builtin_amdgcn_wmma_f32_16x16x32_bf16(false, aq0, false, bk0,
                                                    (short)0, acc, false, false);
      acc = __builtin_amdgcn_wmma_f32_16x16x32_bf16(false, aq1, false, bk1,
                                                    (short)0, acc, false, false);
#pragma unroll
      for (int j = 0; j < 8; ++j) {
        float s    = acc[j] * 0.125f;             // / sqrt(64)
        float seff = cm ? s : -1e30f;
        float nm   = fmaxf(mL[j], seff);
        lL[j] = lL[j] * __expf(mL[j] - nm) + (cm ? __expf(seff - nm) : 0.f);
        mL[j] = nm;
      }
    }
    __syncthreads();                              // readers done before overwrite
  }
  // combine partials across the 16 lanes holding each row
#pragma unroll
  for (int d = 1; d < 16; d <<= 1) {
#pragma unroll
    for (int j = 0; j < 8; ++j) {
      float om = __shfl_xor(mL[j], d, 32);
      float ol = __shfl_xor(lL[j], d, 32);
      float nm = fmaxf(mL[j], om);
      lL[j] = lL[j] * __expf(mL[j] - nm) + ol * __expf(om - nm);
      mL[j] = nm;
    }
  }
  float invL[8];
#pragma unroll
  for (int j = 0; j < 8; ++j) invL[j] = (lL[j] > 0.f) ? (1.f / lL[j]) : 0.f;

  // ---- pass 2: probabilities -> column sums ----
  if (wave == 0) tdm_load_chunk(kpanel, lds0);
  for (int c = 0; c < NCHUNK; ++c) {
    __builtin_amdgcn_s_wait_tensorcnt(0);
    __syncthreads();
    if (wave == 0 && c + 1 < NCHUNK)
      tdm_load_chunk(kpanel + (size_t)(c + 1) * CHUNK_ROWS * A_,
                     ((c + 1) & 1) ? lds1 : lds0);
    const unsigned short* kc = &ksh[c & 1][0];
    for (int tl = 0; tl < CHUNK_T; ++tl) {
      const int ct = c * CHUNK_T + tl;
      const bool cm = mrow[ct * 16 + ln] != 0;
      const unsigned short* krow = kc + (tl * 16 + ln) * A_;
      v16bf bk0, bk1;
      load_kfrag(krow, kb, bk0, bk1);
      v8f acc = {};
      acc = __builtin_amdgcn_wmma_f32_16x16x32_bf16(false, aq0, false, bk0,
                                                    (short)0, acc, false, false);
      acc = __builtin_amdgcn_wmma_f32_16x16x32_bf16(false, aq1, false, bk1,
                                                    (short)0, acc, false, false);
      float ca = 0.f, cw = 0.f;
#pragma unroll
      for (int j = 0; j < 8; ++j) {
        float e = cm ? __expf(fminf(acc[j] * 0.125f - mL[j], 0.f)) : 0.f;
        float p = e * invL[j];
        ca += p;
        cw += mr[j] * p;
      }
      ca += __shfl_xor(ca, 16, 32);
      cw += __shfl_xor(cw, 16, 32);
      if (half == 0) {
        atomicAdd(&w_all[(size_t)b * S_ + ct * 16 + ln], ca);
        atomicAdd(&w_msk[(size_t)b * S_ + ct * 16 + ln], cw);
      }
    }
    __syncthreads();
  }
}

// ---------------------------------------------------------------------------
// small helpers
// ---------------------------------------------------------------------------
__global__ void __launch_bounds__(256)
zero_ws(float* __restrict__ w_all, float* __restrict__ w_msk, float* __restrict__ y) {
  int i = blockIdx.x * 256 + threadIdx.x;
  if (i < B_ * S_) { w_all[i] = 0.f; w_msk[i] = 0.f; }
  if (i < B_ * H_) y[i] = 0.f;
}

__global__ void __launch_bounds__(256)
batch_stats(const int* __restrict__ mask, const float* __restrict__ w_msk,
            float* __restrict__ sumw, float* __restrict__ denom) {
  __shared__ float sm[256], sw[256];
  int b = blockIdx.x, t = threadIdx.x;
  float a = 0.f, c = 0.f;
  for (int s = t; s < S_; s += 256) {
    a += (float)mask[(size_t)b * S_ + s];
    c += w_msk[(size_t)b * S_ + s];
  }
  sm[t] = a; sw[t] = c; __syncthreads();
  for (int st = 128; st > 0; st >>= 1) {
    if (t < st) { sm[t] += sm[t + st]; sw[t] += sw[t + st]; }
    __syncthreads();
  }
  if (t == 0) { denom[b] = sm[0] + 1e-8f; sumw[b] = sw[0]; }
}

// y[b,h] = sum_s w_msk[b,s] * x[b,s,h]
__global__ void __launch_bounds__(256)
wx_gemv(const float* __restrict__ x, const float* __restrict__ w_msk,
        float* __restrict__ y) {
  int h  = blockIdx.x * 256 + threadIdx.x;
  int sc = blockIdx.y;
  int b  = blockIdx.z;
  const float* xb = x + ((size_t)b * S_) * H_;
  float acc = 0.f;
  for (int s = sc * 256; s < sc * 256 + 256; ++s)
    acc = fmaf(w_msk[(size_t)b * S_ + s], xb[(size_t)s * H_ + h], acc);
  atomicAdd(&y[(size_t)b * H_ + h], acc);
}

// out[b,h] = (y[b,:] @ Wv[:,h] + sumw[b]*bv[h]) / denom[b]
__global__ void __launch_bounds__(256)
out_proj(const float* __restrict__ y, const float* __restrict__ Wv,
         const float* __restrict__ bv, const float* __restrict__ sumw,
         const float* __restrict__ denom, float* __restrict__ out) {
  int h = blockIdx.x * 256 + threadIdx.x;
  int b = blockIdx.y;
  float acc = 0.f;
  for (int k = 0; k < H_; ++k)
    acc = fmaf(y[(size_t)b * H_ + k], Wv[(size_t)k * H_ + h], acc);
  out[(size_t)b * H_ + h] = (acc + sumw[b] * bv[h]) / denom[b];
}

__global__ void __launch_bounds__(256)
attn_mean(const float* __restrict__ w_all, float* __restrict__ out2) {
  int i = blockIdx.x * 256 + threadIdx.x;
  if (i < B_ * S_) out2[i] = w_all[i] * (1.0f / (float)S_);
}

// ---------------------------------------------------------------------------
extern "C" void kernel_launch(void* const* d_in, const int* in_sizes, int n_in,
                              void* d_out, int out_size, void* d_ws, size_t ws_size,
                              hipStream_t stream) {
  const float* x    = (const float*)d_in[0];
  const int*   mask = (const int*)  d_in[1];
  const float* Wq   = (const float*)d_in[2];
  const float* bq   = (const float*)d_in[3];
  const float* Wk   = (const float*)d_in[4];
  const float* bk   = (const float*)d_in[5];
  const float* Wv   = (const float*)d_in[6];
  const float* bv   = (const float*)d_in[7];
  float* out  = (float*)d_out;                       // (B,H) = 16384 floats
  float* out2 = out + (size_t)B_ * H_;               // (B,S) = 32768 floats

  char* ws = (char*)d_ws;
  const size_t QB_OFF   = 0;
  const size_t KB_OFF   = QB_OFF   + (size_t)B_ * S_ * A_ * 2;   // bf16 Q
  const size_t WQB_OFF  = KB_OFF   + (size_t)B_ * S_ * A_ * 2;   // bf16 K
  const size_t WKB_OFF  = WQB_OFF  + (size_t)65536 * 2;
  const size_t WALL_OFF = WKB_OFF  + (size_t)65536 * 2;
  const size_t WMSK_OFF = WALL_OFF + (size_t)B_ * S_ * 4;
  const size_t Y_OFF    = WMSK_OFF + (size_t)B_ * S_ * 4;
  const size_t SUMW_OFF = Y_OFF    + (size_t)B_ * H_ * 4;
  const size_t DEN_OFF  = SUMW_OFF + 64;

  unsigned short* Qb   = (unsigned short*)(ws + QB_OFF);
  unsigned short* Kb   = (unsigned short*)(ws + KB_OFF);
  unsigned short* WqB  = (unsigned short*)(ws + WQB_OFF);
  unsigned short* WkB  = (unsigned short*)(ws + WKB_OFF);
  float* w_all = (float*)(ws + WALL_OFF);
  float* w_msk = (float*)(ws + WMSK_OFF);
  float* y     = (float*)(ws + Y_OFF);
  float* sumw  = (float*)(ws + SUMW_OFF);
  float* denom = (float*)(ws + DEN_OFF);

  zero_ws    <<<(B_ * S_ + 255) / 256, 256, 0, stream>>>(w_all, w_msk, y);
  prep_wfrag <<<65536 / 256, 256, 0, stream>>>(Wq, Wk, WqB, WkB);
  proj_qk    <<<(B_ * S_ / 16) / 4, 128, 0, stream>>>(x, bq, bk, WqB, WkB, Qb, Kb);
  attn_colsum<<<(B_ * S_ / 16) / 8, 256, 0, stream>>>(Qb, Kb, mask, w_all, w_msk);
  batch_stats<<<B_, 256, 0, stream>>>(mask, w_msk, sumw, denom);
  wx_gemv    <<<dim3(H_ / 256, 8, B_), 256, 0, stream>>>(x, w_msk, y);
  out_proj   <<<dim3(H_ / 256, B_), 256, 0, stream>>>(y, Wv, bv, sumw, denom, out);
  attn_mean  <<<(B_ * S_ + 255) / 256, 256, 0, stream>>>(w_all, out2);
}
